// Net_63239098466585
// MI455X (gfx1250) — matrix operand order repaired and checked by reading, not compiled, not run on hardware
//
#include <hip/hip_runtime.h>
#include <hip/hip_bf16.h>

typedef __attribute__((ext_vector_type(16))) __bf16 v16bf;
typedef __attribute__((ext_vector_type(8)))  float  v8f;
typedef unsigned short u16;

#define HW     3136   // 56*56
#define CIN    512
#define BATCH  32
#define KMIX   48
#define NCLUT  5

// ---------- helpers ----------
__device__ __forceinline__ u16 f2bf(float f) {
    unsigned u = __float_as_uint(f);
    unsigned r = u + 0x7FFFu + ((u >> 16) & 1u);   // round-to-nearest-even
    return (u16)(r >> 16);
}
__device__ __forceinline__ float clip01(float v) { return fminf(fmaxf(v, 0.f), 1.f); }

union BfOp { int4 i[2]; v16bf v; };

__device__ __forceinline__ v8f vzero8() {
    v8f z = {0.f,0.f,0.f,0.f,0.f,0.f,0.f,0.f};
    return z;
}
__device__ __forceinline__ v8f wmma_bf16(const BfOp& a, const BfOp& b, v8f c) {
    return __builtin_amdgcn_wmma_f32_16x16x32_bf16(
        false, a.v, false, b.v, (short)0, c, false, false);
}

// ---------- prep: normalize conv weights -> bf16, chunk-major [k/32][n(512)][k%32] ----------
__global__ __launch_bounds__(256) void prep_wn(const float* __restrict__ w,
                                               u16* __restrict__ wn) {
    int n = blockIdx.x;           // filter row, 512 blocks
    int t = threadIdx.x;
    __shared__ float red[256];
    float v0 = w[n * CIN + t];
    float v1 = w[n * CIN + t + 256];
    red[t] = v0 * v0 + v1 * v1;
    __syncthreads();
    for (int s = 128; s > 0; s >>= 1) {
        if (t < s) red[t] += red[t + s];
        __syncthreads();
    }
    float norm = sqrtf(red[0]);
    float inv  = norm > 0.f ? 1.f / norm : 0.f;
    int k0 = t, k1 = t + 256;
    wn[(k0 >> 5) * (CIN * 32) + n * 32 + (k0 & 31)] = f2bf(v0 * inv);
    wn[(k1 >> 5) * (CIN * 32) + n * 32 + (k1 & 31)] = f2bf(v1 * inv);
}

// ---------- prep: clutter -> clip, L1-normalize, pad to 16 rows, chunk-major [k/32][n(16)][k%32] ----------
__global__ __launch_bounds__(256) void prep_ck(const float* __restrict__ clut,
                                               u16* __restrict__ ck) {
    int t = threadIdx.x;
    __shared__ float s5[NCLUT];
    if (t < NCLUT) {
        float s = 0.f;
        for (int k = 0; k < CIN; ++k) s += clip01(clut[t * CIN + k]);
        s5[t] = fmaxf(s, 1e-12f);
    }
    __syncthreads();
    for (int idx = t; idx < 16 * CIN; idx += 256) {
        int n = idx >> 9;        // 0..15
        int k = idx & 511;
        float v = 0.f;
        if (n < NCLUT) v = clip01(clut[n * CIN + k]) / s5[n];
        ck[(k >> 5) * (16 * 32) + n * 32 + (k & 31)] = f2bf(v);
    }
}

__global__ void zero_accum(float* __restrict__ acc) {
    for (int i = threadIdx.x; i < BATCH * KMIX; i += 256) acc[i] = 0.f;
}

// ---------- main fused kernel: one pixel per workgroup ----------
__global__ __launch_bounds__(256) void vmf_pixel(const float* __restrict__ vgg,
                                                 const float* __restrict__ mix,
                                                 const u16*  __restrict__ wn,
                                                 const u16*  __restrict__ ck,
                                                 float*      __restrict__ accum) {
    extern __shared__ char smem[];
    u16*   x_bf   = (u16*)(smem);                  // [32][512] bf16   32768 B
    u16*   act_bf = (u16*)(smem + 32768);          // [32][512] bf16   32768 B
    u16*   mm_bf  = (u16*)(smem + 65536);          // [48][512] bf16   49152 B
    float* xss    = (float*)(smem + 114688);       // [32]
    float* invx   = (float*)(smem + 114816);       // [32]
    float* msum   = (float*)(smem + 114944);       // [48]
    float* minv   = (float*)(smem + 115136);       // [48]
    float* bgtmp  = (float*)(smem + 115328);       // [32][16]          2048 B
    float* bgrow  = (float*)(smem + 117376);       // [32]
    // total 117504 B dynamic LDS

    const int pix  = blockIdx.x;                   // 0..3135
    const int t    = threadIdx.x;
    const int wave = t >> 5;
    const int lane = t & 31;
    const int lo   = lane & 15;
    const int hi   = lane >> 4;

    if (t < BATCH) xss[t]  = 0.f;
    if (t < KMIX)  msum[t] = 0.f;
    __syncthreads();

    // ---- stage x slice (gather, compute ||x||^2 with ds_add_f32) ----
    {
        int b  = t >> 3;               // 8 threads per batch row
        int k0 = (t & 7) * 64;
        const float* src = vgg + (size_t)b * (CIN * HW) + pix;
        float ss = 0.f;
        for (int kk = 0; kk < 64; ++kk) {
            int k = k0 + kk;
            float v = src[(size_t)k * HW];
            ss += v * v;
            x_bf[b * CIN + k] = f2bf(v);
        }
        atomicAdd(&xss[b], ss);
    }
    // ---- stage mm slice (clip, keep unnormalized; L1 sum via ds_add_f32) ----
    for (int idx = t; idx < KMIX * 64; idx += 256) {
        int m  = idx >> 6;
        int k0 = (idx & 63) * 8;
        const float* src = mix + (size_t)m * (CIN * HW) + pix;
        float ps = 0.f;
        for (int r = 0; r < 8; ++r) {
            int k = k0 + r;
            float v = clip01(src[(size_t)k * HW]);
            ps += v;
            mm_bf[m * CIN + k] = f2bf(v);
        }
        atomicAdd(&msum[m], ps);
    }
    __syncthreads();
    if (t < BATCH) {
        float norm = sqrtf(xss[t]);
        invx[t] = norm > 0.f ? 1.f / norm : 1.f;   // matches xnorm += (xnorm==0)
    }
    if (t < KMIX) minv[t] = 1.f / fmaxf(msum[t], 1e-12f);
    __syncthreads();

    // ---- GEMM1: cos(32x512) = x_bf(32x512) * wn^T ; wave owns 2 M-tiles x 4 N-tiles ----
    v8f acc1[2][4];
    for (int m = 0; m < 2; ++m)
        for (int j = 0; j < 4; ++j) acc1[m][j] = vzero8();

    for (int kc = 0; kc < 16; ++kc) {
        BfOp A[2];
#pragma unroll
        for (int mt = 0; mt < 2; ++mt) {
            const u16* p = x_bf + (mt * 16 + lo) * CIN + kc * 32 + hi * 16;
            A[mt].i[0] = *(const int4*)(p);
            A[mt].i[1] = *(const int4*)(p + 8);
        }
#pragma unroll
        for (int j = 0; j < 4; ++j) {
            int n = (wave * 4 + j) * 16 + lo;
            const u16* bp = wn + kc * (CIN * 32) + n * 32 + hi * 16;
            BfOp B;
            B.i[0] = *(const int4*)(bp);
            B.i[1] = *(const int4*)(bp + 8);
            acc1[0][j] = wmma_bf16(A[0], B, acc1[0][j]);
            acc1[1][j] = wmma_bf16(A[1], B, acc1[1][j]);
        }
    }
    // activation: act = exp(30*cos) * (cos > 0), store bf16 to LDS
#pragma unroll
    for (int mt = 0; mt < 2; ++mt)
#pragma unroll
        for (int j = 0; j < 4; ++j) {
            int colg = (wave * 4 + j) * 16 + lo;
#pragma unroll
            for (int i = 0; i < 8; ++i) {
                int row  = mt * 16 + i + 8 * hi;
                float cs = acc1[mt][j][i] * invx[row];
                float a  = (cs > 0.f) ? __expf(30.f * cs) : 0.f;
                act_bf[row * CIN + colg] = f2bf(a);
            }
        }
    __syncthreads();

    // ---- GEMM2: fg(32x48) on waves 0..5 ; bg(32x16) on waves 6..7 ----
    const bool isfg = wave < 6;
    const int  mt2  = isfg ? (wave & 1) : (wave - 6);
    const int  nt2  = isfg ? (wave >> 1) : 0;

    v8f o = vzero8();
    for (int kc = 0; kc < 16; ++kc) {
        BfOp A;
        {
            const u16* p = act_bf + (mt2 * 16 + lo) * CIN + kc * 32 + hi * 16;
            A.i[0] = *(const int4*)(p);
            A.i[1] = *(const int4*)(p + 8);
        }
        BfOp B;
        if (isfg) {
            const u16* bp = mm_bf + (nt2 * 16 + lo) * CIN + kc * 32 + hi * 16;
            B.i[0] = *(const int4*)(bp);
            B.i[1] = *(const int4*)(bp + 8);
        } else {
            const u16* bp = ck + kc * (16 * 32) + lo * 32 + hi * 16;
            B.i[0] = *(const int4*)(bp);
            B.i[1] = *(const int4*)(bp + 8);
        }
        o = wmma_bf16(A, B, o);
    }

    // bg epilogue: background = max_n log(bg*0.6 + 1e-10)
    if (!isfg) {
#pragma unroll
        for (int i = 0; i < 8; ++i) {
            int row = mt2 * 16 + i + 8 * hi;
            float v = (lo < NCLUT) ? __logf(o[i] * 0.6f + 1e-10f) : -3.0e38f;
            bgtmp[row * 16 + lo] = v;
        }
    }
    __syncthreads();
    if (t < BATCH) {
        float m = -3.0e38f;
        for (int j = 0; j < 16; ++j) m = fmaxf(m, bgtmp[t * 16 + j]);
        bgrow[t] = m;
    }
    __syncthreads();

    // fg epilogue: fold mm L1 scale, log, max with background, accumulate over pixels
    if (isfg) {
        int colg = nt2 * 16 + lo;
        float sc = minv[colg];
#pragma unroll
        for (int i = 0; i < 8; ++i) {
            int row = mt2 * 16 + i + 8 * hi;
            float v = __logf(o[i] * sc * 0.4f + 1e-10f);
            float c = fmaxf(v, bgrow[row]);
            atomicAdd(accum + row * KMIX + colg, c);
        }
    }
}

// ---------- finalize: mixture max, softmax ----------
__global__ __launch_bounds__(384) void finalize(const float* __restrict__ accum,
                                                float* __restrict__ soft_out,
                                                float* __restrict__ ml_out) {
    __shared__ float zsh[BATCH][12];
    __shared__ float mlsh[BATCH][12];
    int t = threadIdx.x;
    if (t < BATCH * 12) {
        int b = t / 12, c = t % 12;
        float m = -3.0e38f;
        for (int j = 0; j < 4; ++j) m = fmaxf(m, accum[b * KMIX + c * 4 + j]);
        float ml = m / (float)HW;
        mlsh[b][c] = ml;
        float e = fminf(fmaxf(ml * 2.f, -88.7f), 88.7f);
        zsh[b][c] = __expf(e);
    }
    __syncthreads();
    if (t < BATCH * 12) {
        int b = t / 12, c = t % 12;
        float s = 0.f;
        for (int cc = 0; cc < 12; ++cc) s += zsh[b][cc];
        soft_out[t] = zsh[b][c] / s;
        ml_out[t]   = mlsh[b][c];
    }
}

// ---------- passthrough copy of vgg_feat ----------
__global__ void copy_f4(const float4* __restrict__ src, float4* __restrict__ dst, int n) {
    for (int i = blockIdx.x * blockDim.x + threadIdx.x; i < n;
         i += gridDim.x * blockDim.x)
        dst[i] = src[i];
}

extern "C" void kernel_launch(void* const* d_in, const int* in_sizes, int n_in,
                              void* d_out, int out_size, void* d_ws, size_t ws_size,
                              hipStream_t stream) {
    const float* vgg   = (const float*)d_in[0];   // (32,512,56,56)
    const float* convw = (const float*)d_in[1];   // (512,512,1,1)
    const float* mix   = (const float*)d_in[2];   // (48,512,56,56)
    const float* clut  = (const float*)d_in[3];   // (5,512)

    u16*   ws_wn = (u16*)d_ws;                             // 524288 B
    u16*   ws_ck = (u16*)((char*)d_ws + 524288);           //  16384 B
    float* accum = (float*)((char*)d_ws + 540672);         //   6144 B

    float* out = (float*)d_out;
    const int vgg_elems = BATCH * CIN * HW;                // 51380224
    float* soft_out = out;
    float* vgg_out  = out + 384;
    float* ml_out   = out + 384 + vgg_elems;

    prep_wn<<<512, 256, 0, stream>>>(convw, ws_wn);
    prep_ck<<<1, 256, 0, stream>>>(clut, ws_ck);
    zero_accum<<<1, 256, 0, stream>>>(accum);

    vmf_pixel<<<HW, 256, 117504, stream>>>(vgg, mix, ws_wn, ws_ck, accum);

    finalize<<<1, 384, 0, stream>>>(accum, soft_out, ml_out);
    copy_f4<<<6272, 256, 0, stream>>>((const float4*)vgg, (float4*)vgg_out,
                                      vgg_elems / 4);
}
